// LTAttention_16252156248654
// MI455X (gfx1250) — matrix operand order repaired and checked by reading, not compiled, not run on hardware
//
#include <hip/hip_runtime.h>
#include <hip/hip_bf16.h>

// ---------------------------------------------------------------------------
// Problem constants (from reference): B=2, S=1536, H=2048, NH=16, DH=128
// ---------------------------------------------------------------------------
#define BATCH 2
#define SEQ   1536
#define HID   2048
#define NHEAD 16
#define DHEAD 128
#define MROWS (BATCH * SEQ)             // 3072
#define FP16_MIN_F (-65504.0f)
#define SM_SCALE 0.08838834764831845f   // 1/sqrt(128)

typedef __attribute__((ext_vector_type(16))) _Float16 v16h;
typedef __attribute__((ext_vector_type(8)))  _Float16 v8h;
typedef __attribute__((ext_vector_type(8)))  float    v8f;
typedef __attribute__((ext_vector_type(4)))  float    v4f;
typedef __attribute__((ext_vector_type(4)))  int      v4i;

// ---------------------------------------------------------------------------
// CDNA5 async global->LDS copy (ASYNCcnt-tracked), with sync fallback.
// ---------------------------------------------------------------------------
#if __has_builtin(__builtin_amdgcn_global_load_async_to_lds_b128) && \
    __has_builtin(__builtin_amdgcn_s_wait_asynccnt)
#define USE_ASYNC_LDS 1
#else
#define USE_ASYNC_LDS 0
#endif

__device__ __forceinline__ void lds_copy16B(_Float16* lds_dst, const _Float16* gsrc) {
#if USE_ASYNC_LDS
  __builtin_amdgcn_global_load_async_to_lds_b128(
      (__attribute__((address_space(1))) v4i*)(void*)gsrc,
      (__attribute__((address_space(3))) v4i*)(void*)lds_dst,
      0, 0);
#else
  *(v8h*)lds_dst = *(const v8h*)gsrc;
#endif
}

__device__ __forceinline__ void lds_copy_wait() {
#if USE_ASYNC_LDS
  __builtin_amdgcn_s_wait_asynccnt(0);
#endif
}

// ---------------------------------------------------------------------------
// Fragment loaders for V_WMMA_F32_16X16X32_F16 (wave32 layouts per ISA 7.12.2)
// A (16x32): lane l -> row l&15; khalf=(l>>4)*8; VGPR0..3 K=khalf+0..7,
//            VGPR4..7 K=16+khalf+0..7.
// B (32x16): lane l -> col l&15; khalf=(l>>4)*8; VGPR0..3 K=khalf+0..7,
//            VGPR4..7 K=16+khalf+0..7.
// C/D (16x16 f32): lane l -> col l&15; VGPR r -> row r+8*(l>>4).
// ---------------------------------------------------------------------------
__device__ __forceinline__ v16h load_a_frag16(const _Float16* __restrict__ base,
                                              int ld, int m0, int kc, int lane) {
  const _Float16* p = base + (size_t)(m0 + (lane & 15)) * ld + kc + ((lane >> 4) << 3);
  union { v16h v; v8h h[2]; } u;
  u.h[0] = *(const v8h*)(p);
  u.h[1] = *(const v8h*)(p + 16);
  return u.v;
}

// B fragment from row-major [N, K] storage (row n supplies output column n).
__device__ __forceinline__ v16h load_b_frag16(const _Float16* base,
                                              int ld, int n0, int kc, int lane) {
  const _Float16* p = base + (size_t)(n0 + (lane & 15)) * ld + kc + ((lane >> 4) << 3);
  union { v16h v; v8h h[2]; } u;
  u.h[0] = *(const v8h*)(p);
  u.h[1] = *(const v8h*)(p + 16);
  return u.v;
}

// ---------------------------------------------------------------------------
// f32 -> f16 conversion (one-time); single-use f32 stream -> non-temporal
// ---------------------------------------------------------------------------
__global__ __launch_bounds__(256)
void cvt_f32_f16_kernel(const float* __restrict__ in, _Float16* __restrict__ out, int n) {
  const int i = (blockIdx.x * 256 + threadIdx.x) * 8;
  if (i + 8 > n) return;
  v4f a = __builtin_nontemporal_load((const v4f*)(in + i));
  v4f b = __builtin_nontemporal_load((const v4f*)(in + i + 4));
  v8h o;
  o[0] = (_Float16)a[0]; o[1] = (_Float16)a[1]; o[2] = (_Float16)a[2]; o[3] = (_Float16)a[3];
  o[4] = (_Float16)b[0]; o[5] = (_Float16)b[1]; o[6] = (_Float16)b[2]; o[7] = (_Float16)b[3];
  *(v8h*)(out + i) = o;
}

// ---------------------------------------------------------------------------
// GEMM: C[M,N] = A[M,K] @ W[N,K]^T, f16 in.
// MODE 0: f32 out [M,N] (non-temporal final output);
// MODE 1: f16 out [M,N]; MODE 2: f16 out transposed to V^T [b, h, dh, s].
// Block = 128 threads (4 waves). Block tile 128(M) x 64(N); 8 WMMA/k-step.
// W tile (64x32 f16 = 4KB) staged in LDS, double buffered, async-filled.
// ---------------------------------------------------------------------------
template <int MODE>
__global__ __launch_bounds__(128)
void gemm_nt_wmma_f16(const _Float16* __restrict__ A,
                      const _Float16* __restrict__ W,
                      void* __restrict__ Cout,
                      int M, int N, int K) {
  __shared__ _Float16 Bt[2][64 * 32];

  const int lane = threadIdx.x & 31;
  const int wave = threadIdx.x >> 5;
  const int m0 = blockIdx.x * 128 + wave * 32;
  const int n0 = blockIdx.y * 64;

  const int sn = threadIdx.x >> 1;
  const int sk = (threadIdx.x & 1) << 4;
  const _Float16* wrow = W + (size_t)(n0 + sn) * K + sk;
  _Float16* lrow0 = &Bt[0][sn * 32 + sk];
  _Float16* lrow1 = &Bt[1][sn * 32 + sk];

  v8f acc[2][4];
  #pragma unroll
  for (int i = 0; i < 2; ++i)
    #pragma unroll
    for (int j = 0; j < 4; ++j) acc[i][j] = (v8f){};

  lds_copy16B(lrow0, wrow);
  lds_copy16B(lrow0 + 8, wrow + 8);
  lds_copy_wait();
  __syncthreads();

  int buf = 0;
  for (int kc = 0; kc < K; kc += 32) {
    if (kc + 32 < K) {
      _Float16* dst = buf ? lrow0 : lrow1;
      const _Float16* src = wrow + kc + 32;
      lds_copy16B(dst, src);
      lds_copy16B(dst + 8, src + 8);
    }

    v16h a0 = load_a_frag16(A, K, m0, kc, lane);
    v16h a1 = load_a_frag16(A, K, m0 + 16, kc, lane);
    const _Float16* tile = &Bt[buf][0];
    #pragma unroll
    for (int nt = 0; nt < 4; ++nt) {
      v16h bfr = load_b_frag16(tile, 32, nt * 16, 0, lane);
      acc[0][nt] = __builtin_amdgcn_wmma_f32_16x16x32_f16(false, a0, false, bfr, (short)0, acc[0][nt], false, false);
      acc[1][nt] = __builtin_amdgcn_wmma_f32_16x16x32_f16(false, a1, false, bfr, (short)0, acc[1][nt], false, false);
    }

    lds_copy_wait();
    __syncthreads();
    buf ^= 1;
  }

  const int coln  = lane & 15;
  const int rbase = (lane >> 4) << 3;
  #pragma unroll
  for (int half = 0; half < 2; ++half) {
    #pragma unroll
    for (int r = 0; r < 8; ++r) {
      const int mrow = m0 + half * 16 + rbase + r;
      if (MODE == 2) {
        // V^T layout: [b, h, dh, s], s fastest
        const int bb = mrow / SEQ, ss = mrow % SEQ;
        _Float16* c = (_Float16*)Cout;
        #pragma unroll
        for (int nt = 0; nt < 4; ++nt) {
          const int n = n0 + nt * 16 + coln;
          const int hh = n >> 7, dh = n & 127;
          c[((size_t)(bb * NHEAD + hh) * DHEAD + dh) * SEQ + ss] = (_Float16)acc[half][nt][r];
        }
      } else {
        const size_t row = (size_t)mrow * N + n0 + coln;
        if (MODE == 1) {
          _Float16* c = (_Float16*)Cout;
          #pragma unroll
          for (int nt = 0; nt < 4; ++nt) c[row + nt * 16] = (_Float16)acc[half][nt][r];
        } else {
          float* c = (float*)Cout;
          #pragma unroll
          for (int nt = 0; nt < 4; ++nt)
            __builtin_nontemporal_store(acc[half][nt][r], &c[row + nt * 16]);
        }
      }
    }
  }
}

// ---------------------------------------------------------------------------
// Flash attention: block = 128 threads (4 waves) handles 64 query rows of one
// (b,h). Per 32-key step: K tile (32x128, 8KB) and V^T tile (128x32, 8KB)
// async-staged to LDS double-buffered; each wave runs 8 score WMMAs (K-dim
// 128), online softmax, and 8 PV WMMAs from LDS fragments. Output f16 [B*S,H].
// Bias (302MB, single-use, > L2) is streamed non-temporally.
// ---------------------------------------------------------------------------
__global__ __launch_bounds__(128)
void flash_attn_wmma_kernel(const _Float16* __restrict__ Q,
                            const _Float16* __restrict__ Km,
                            const _Float16* __restrict__ Vt,
                            const float* __restrict__ bias,
                            _Float16* __restrict__ O) {
  __shared__ _Float16 KtS[2][32 * DHEAD];   // [key][dh]
  __shared__ _Float16 VtS[2][DHEAD * 32];   // [dh][key]
  __shared__ _Float16 Pt[4][16 * 32];       // per-wave P tiles

  const int t    = threadIdx.x;
  const int lane = t & 31;
  const int wave = t >> 5;
  const int bh   = blockIdx.x;
  const int b    = bh / NHEAD;
  const int h    = bh % NHEAD;
  const int q0b  = blockIdx.y * 64;         // block query base
  const int q0   = q0b + wave * 16;         // this wave's query base

  const _Float16* qp  = Q + (size_t)b * SEQ * HID + h * DHEAD;
  const _Float16* kp  = Km + (size_t)b * SEQ * HID + h * DHEAD;
  const _Float16* vtp = Vt + (size_t)bh * DHEAD * SEQ;
  const float*    bp  = bias + (size_t)bh * SEQ * SEQ;

  // staging slices for this thread
  const int krow   = t >> 2;          // 0..31
  const int kchunk = (t & 3) << 5;    // 0,32,64,96 halves (64B)

  v16h qf0 = load_a_frag16(qp, HID, q0, 0,  lane);
  v16h qf1 = load_a_frag16(qp, HID, q0, 32, lane);
  v16h qf2 = load_a_frag16(qp, HID, q0, 64, lane);
  v16h qf3 = load_a_frag16(qp, HID, q0, 96, lane);

  v8f acc[8];
  #pragma unroll
  for (int nt = 0; nt < 8; ++nt) acc[nt] = (v8f){};
  float mrow[8], lrow[8];
  #pragma unroll
  for (int r = 0; r < 8; ++r) { mrow[r] = -3.0e38f; lrow[r] = 0.0f; }

  const int coln  = lane & 15;
  const int rhalf = (lane >> 4) << 3;
  const int jmaxb = (q0b + 64 + 31) >> 5;   // key blocks covering the whole block

  // ---- prime buffer 0 with key block 0 ----
  {
    const _Float16* ks = kp + (size_t)krow * HID + kchunk;
    _Float16* kd = &KtS[0][krow * DHEAD + kchunk];
    lds_copy16B(kd, ks); lds_copy16B(kd + 8, ks + 8);
    lds_copy16B(kd + 16, ks + 16); lds_copy16B(kd + 24, ks + 24);
    const _Float16* vs = vtp + (size_t)t * SEQ;
    _Float16* vd = &VtS[0][t * 32];
    lds_copy16B(vd, vs); lds_copy16B(vd + 8, vs + 8);
    lds_copy16B(vd + 16, vs + 16); lds_copy16B(vd + 24, vs + 24);
  }
  lds_copy_wait();
  __syncthreads();

  int buf = 0;
  for (int j = 0; j < jmaxb; ++j) {
    const int k0 = j * 32;

    // ---- stage next key block into the other buffer ----
    if (j + 1 < jmaxb) {
      const int kn = k0 + 32;
      const _Float16* ks = kp + (size_t)(kn + krow) * HID + kchunk;
      _Float16* kd = &KtS[buf ^ 1][krow * DHEAD + kchunk];
      lds_copy16B(kd, ks); lds_copy16B(kd + 8, ks + 8);
      lds_copy16B(kd + 16, ks + 16); lds_copy16B(kd + 24, ks + 24);
      const _Float16* vs = vtp + (size_t)t * SEQ + kn;
      _Float16* vd = &VtS[buf ^ 1][t * 32];
      lds_copy16B(vd, vs); lds_copy16B(vd + 8, vs + 8);
      lds_copy16B(vd + 16, vs + 16); lds_copy16B(vd + 24, vs + 24);
    }

    if (k0 < q0 + 16) {   // causal: this wave still has unmasked keys here
      const _Float16* ktile = &KtS[buf][0];
      const _Float16* vtile = &VtS[buf][0];

      // ---- scores: S = Q @ K^T (two 16x16 tiles, contraction over DH=128) ----
      v8f s0 = {}, s1 = {};
      {
        v16h kb0, kb1;
        kb0 = load_b_frag16(ktile, DHEAD, 0,  0,  lane);
        kb1 = load_b_frag16(ktile, DHEAD, 16, 0,  lane);
        s0 = __builtin_amdgcn_wmma_f32_16x16x32_f16(false, qf0, false, kb0, (short)0, s0, false, false);
        s1 = __builtin_amdgcn_wmma_f32_16x16x32_f16(false, qf0, false, kb1, (short)0, s1, false, false);
        kb0 = load_b_frag16(ktile, DHEAD, 0,  32, lane);
        kb1 = load_b_frag16(ktile, DHEAD, 16, 32, lane);
        s0 = __builtin_amdgcn_wmma_f32_16x16x32_f16(false, qf1, false, kb0, (short)0, s0, false, false);
        s1 = __builtin_amdgcn_wmma_f32_16x16x32_f16(false, qf1, false, kb1, (short)0, s1, false, false);
        kb0 = load_b_frag16(ktile, DHEAD, 0,  64, lane);
        kb1 = load_b_frag16(ktile, DHEAD, 16, 64, lane);
        s0 = __builtin_amdgcn_wmma_f32_16x16x32_f16(false, qf2, false, kb0, (short)0, s0, false, false);
        s1 = __builtin_amdgcn_wmma_f32_16x16x32_f16(false, qf2, false, kb1, (short)0, s1, false, false);
        kb0 = load_b_frag16(ktile, DHEAD, 0,  96, lane);
        kb1 = load_b_frag16(ktile, DHEAD, 16, 96, lane);
        s0 = __builtin_amdgcn_wmma_f32_16x16x32_f16(false, qf3, false, kb0, (short)0, s0, false, false);
        s1 = __builtin_amdgcn_wmma_f32_16x16x32_f16(false, qf3, false, kb1, (short)0, s1, false, false);
      }

      // ---- scale + bias (non-temporal) + causal mask + online softmax ----
      const int kc0 = k0 + coln;
      const int kc1 = kc0 + 16;
      #pragma unroll
      for (int r = 0; r < 8; ++r) {
        const int qr = q0 + rhalf + r;
        float sc0 = (kc0 <= qr)
            ? (s0[r] * SM_SCALE + __builtin_nontemporal_load(&bp[(size_t)qr * SEQ + kc0]))
            : FP16_MIN_F;
        float sc1 = (kc1 <= qr)
            ? (s1[r] * SM_SCALE + __builtin_nontemporal_load(&bp[(size_t)qr * SEQ + kc1]))
            : FP16_MIN_F;

        float mx = fmaxf(sc0, sc1);
        #pragma unroll
        for (int off = 1; off < 16; off <<= 1)
          mx = fmaxf(mx, __shfl_xor(mx, off, 32));

        const float mnew = fmaxf(mrow[r], mx);
        const float corr = __expf(mrow[r] - mnew);
        const float p0   = __expf(sc0 - mnew);
        const float p1   = __expf(sc1 - mnew);

        float rs = p0 + p1;
        #pragma unroll
        for (int off = 1; off < 16; off <<= 1)
          rs += __shfl_xor(rs, off, 32);

        lrow[r] = lrow[r] * corr + rs;
        mrow[r] = mnew;
        #pragma unroll
        for (int nt = 0; nt < 8; ++nt) acc[nt][r] *= corr;

        Pt[wave][(rhalf + r) * 32 + coln]      = (_Float16)p0;
        Pt[wave][(rhalf + r) * 32 + coln + 16] = (_Float16)p1;
      }

      // ---- reload P as A-fragment (wave-local LDS; DS ops are in-order) ----
      v16h pa;
      {
        const int m     = lane & 15;
        const int khalf = (lane >> 4) << 3;
        union { v16h v; v8h hh[2]; } u;
        u.hh[0] = *(const v8h*)&Pt[wave][m * 32 + khalf];
        u.hh[1] = *(const v8h*)&Pt[wave][m * 32 + 16 + khalf];
        pa = u.v;
      }

      // ---- O += P @ V  (V^T tile: [dh][key], contraction contiguous) ----
      #pragma unroll
      for (int nt = 0; nt < 8; ++nt) {
        v16h vb = load_b_frag16(vtile, 32, nt * 16, 0, lane);
        acc[nt] = __builtin_amdgcn_wmma_f32_16x16x32_f16(false, pa, false, vb, (short)0, acc[nt], false, false);
      }
    }

    lds_copy_wait();      // own async stores for next tile done
    __syncthreads();      // block done reading current buffers
    buf ^= 1;
  }

  // ---- normalize and write out ----
  _Float16* op = O + (size_t)b * SEQ * HID + h * DHEAD;
  #pragma unroll
  for (int r = 0; r < 8; ++r) {
    const float inv = 1.0f / lrow[r];
    const size_t row = (size_t)(q0 + rhalf + r) * HID;
    #pragma unroll
    for (int nt = 0; nt < 8; ++nt)
      op[row + nt * 16 + coln] = (_Float16)(acc[nt][r] * inv);
  }
}

// ---------------------------------------------------------------------------
// Launch: cvt(x, W*) -> Q,K GEMMs, V GEMM (transposed out) -> attention -> out GEMM
// ---------------------------------------------------------------------------
extern "C" void kernel_launch(void* const* d_in, const int* in_sizes, int n_in,
                              void* d_out, int out_size, void* d_ws, size_t ws_size,
                              hipStream_t stream) {
  const float* x    = (const float*)d_in[0];
  const float* bias = (const float*)d_in[1];
  const float* Wq   = (const float*)d_in[2];
  const float* Wk   = (const float*)d_in[3];
  const float* Wv   = (const float*)d_in[4];
  const float* Wo   = (const float*)d_in[5];
  float* out = (float*)d_out;

  const size_t nAct = (size_t)MROWS * HID;   // 6,291,456
  const size_t nWgt = (size_t)HID * HID;     // 4,194,304

  _Float16* xh  = (_Float16*)d_ws;
  _Float16* wqh = xh + nAct;
  _Float16* wkh = wqh + nWgt;
  _Float16* wvh = wkh + nWgt;
  _Float16* woh = wvh + nWgt;
  _Float16* qh  = woh + nWgt;
  _Float16* kh  = qh + nAct;
  _Float16* vth = kh + nAct;   // V^T: [b, h, dh, s], nAct elements
  _Float16* oh  = vth + nAct;
  const size_t needed = (5 * nAct + 4 * nWgt) * sizeof(_Float16);
  if (ws_size < needed) return;

  cvt_f32_f16_kernel<<<dim3(nAct / 2048), dim3(256), 0, stream>>>(x,  xh,  (int)nAct);
  cvt_f32_f16_kernel<<<dim3(nWgt / 2048), dim3(256), 0, stream>>>(Wq, wqh, (int)nWgt);
  cvt_f32_f16_kernel<<<dim3(nWgt / 2048), dim3(256), 0, stream>>>(Wk, wkh, (int)nWgt);
  cvt_f32_f16_kernel<<<dim3(nWgt / 2048), dim3(256), 0, stream>>>(Wv, wvh, (int)nWgt);
  cvt_f32_f16_kernel<<<dim3(nWgt / 2048), dim3(256), 0, stream>>>(Wo, woh, (int)nWgt);

  dim3 ggrid(MROWS / 128, HID / 64);   // 24 x 32
  dim3 gblk(128);
  gemm_nt_wmma_f16<1><<<ggrid, gblk, 0, stream>>>(xh, wqh, qh,  MROWS, HID, HID);
  gemm_nt_wmma_f16<1><<<ggrid, gblk, 0, stream>>>(xh, wkh, kh,  MROWS, HID, HID);
  gemm_nt_wmma_f16<2><<<ggrid, gblk, 0, stream>>>(xh, wvh, vth, MROWS, HID, HID);

  dim3 agrid(BATCH * NHEAD, SEQ / 64);   // 32 x 24
  dim3 ablk(128);
  flash_attn_wmma_kernel<<<agrid, ablk, 0, stream>>>(qh, kh, vth, bias, oh);

  gemm_nt_wmma_f16<0><<<ggrid, gblk, 0, stream>>>(oh, woh, out, MROWS, HID, HID);
}
